// construct_adj4_87093346828657
// MI455X (gfx1250) — matrix operand order repaired and checked by reading, not compiled, not run on hardware
//
#include <hip/hip_runtime.h>
#include <hip/hip_bf16.h>
#include <math.h>

// Problem constants (match reference)
#define BB 8
#define CC 32
#define NN 1024
#define SS 12

typedef __attribute__((ext_vector_type(16))) _Float16 v16h;
typedef __attribute__((ext_vector_type(8)))  _Float16 v8h;
typedef __attribute__((ext_vector_type(8)))  float    v8f;

union FragU { v16h v; v8h h[2]; };

// ---------------------------------------------------------------------------
// Kernel 1: x1 = tanh(W1·x + b1), x2 = tanh(W2·x + b2)
// Input  x : [B, C, N, S] f32
// Output X1,X2 : [B, S, N, C] f16  (row = 32 contiguous halves = 64 bytes)
// ---------------------------------------------------------------------------
__global__ __launch_bounds__(256) void project_tanh_kernel(
    const float* __restrict__ x,
    const float* __restrict__ W1, const float* __restrict__ b1,
    const float* __restrict__ W2, const float* __restrict__ b2,
    _Float16* __restrict__ X1, _Float16* __restrict__ X2)
{
    __shared__ float W1s[CC * CC];
    __shared__ float W2s[CC * CC];
    __shared__ float b1s[CC];
    __shared__ float b2s[CC];

    const int tid = threadIdx.x;
    for (int i = tid; i < CC * CC; i += 256) { W1s[i] = W1[i]; W2s[i] = W2[i]; }
    if (tid < CC) { b1s[tid] = b1[tid]; b2s[tid] = b2[tid]; }
    __syncthreads();

    const int p  = blockIdx.x * 256 + tid;       // 0 .. B*N*S-1
    const int b  = p / (NN * SS);
    const int ns = p % (NN * SS);
    const int n  = ns / SS;
    const int s  = ns % SS;

    // gather 32 channels (stride N*S floats between channels)
    const float* xb = x + (size_t)b * CC * NN * SS + ns;
    float xc[CC];
#pragma unroll
    for (int c = 0; c < CC; ++c) xc[c] = xb[(size_t)c * (NN * SS)];

    _Float16* o1 = X1 + (((size_t)b * SS + s) * NN + n) * CC;
    _Float16* o2 = X2 + (((size_t)b * SS + s) * NN + n) * CC;

#pragma unroll 4
    for (int d = 0; d < CC; ++d) {
        float a1 = b1s[d];
        float a2 = b2s[d];
        const float* w1r = &W1s[d * CC];
        const float* w2r = &W2s[d * CC];
#pragma unroll
        for (int c = 0; c < CC; ++c) {
            a1 = fmaf(w1r[c], xc[c], a1);
            a2 = fmaf(w2r[c], xc[c], a2);
        }
        o1[d] = (_Float16)tanhf(a1);
        o2[d] = (_Float16)tanhf(a2);
    }
}

// ---------------------------------------------------------------------------
// Kernel 2: per (b, s, n-tile of 16):
//   logits[16][1024] = X1_tile^T(C) x X2  via v_wmma_f32_16x16x32_f16
//   softmax over k, write adj[b, n, k, s]
// Block = 128 threads (4 waves, wave32). LDS = 16*1024 f32 = 64 KB.
// ---------------------------------------------------------------------------
__global__ __launch_bounds__(128) void gemm_softmax_kernel(
    const _Float16* __restrict__ X1,
    const _Float16* __restrict__ X2,
    float* __restrict__ out)
{
    __shared__ float logits[16 * NN];   // 64 KB (WGP LDS is 320 KB)

    const int n_t  = blockIdx.x;        // 0..63  n-tile
    const int s    = blockIdx.y;        // 0..11
    const int b    = blockIdx.z;        // 0..7
    const int tid  = threadIdx.x;
    const int lane = tid & 31;
    const int wv   = tid >> 5;          // 0..3
    const int hsel = lane >> 4;         // lane half: 0 or 1
    const int lrow = lane & 15;

    const size_t slice = ((size_t)b * SS + s) * NN * CC;  // in halves
    const int n0 = n_t * 16;

    // ---- A fragment (16x32 f16, loop-invariant): A[m,c] = X1[.., n0+m, c]
    // ISA layout: lanes 0-15 hold K=0..7 (vgpr0-3) and K=16..23 (vgpr4-7),
    //             lanes 16-31 hold K=8..15 and K=24..31, row M = lane%16.
    const v8h* arow = (const v8h*)(X1 + slice + (size_t)(n0 + lrow) * CC);
    FragU afrag;
    afrag.h[0] = arow[hsel];        // c chunk 0..7  (or 8..15)
    afrag.h[1] = arow[hsel + 2];    // c chunk 16..23 (or 24..31)

    // ---- 64 k-tiles, 16 per wave
    for (int kt = wv; kt < NN / 16; kt += 4) {
        const int k0 = kt * 16;

        // B fragment (32x16 f16): B[c,nn] = X2[.., k0+nn, c]
        // ISA layout: lanes 0-15 hold K(=c)=0..7 and 16..23 of column nn=lane,
        //             lanes 16-31 hold K=8..15 and 24..31 of column nn=lane-16.
        const v8h* brow = (const v8h*)(X2 + slice + (size_t)(k0 + lrow) * CC);
        FragU bfrag;
        bfrag.h[0] = brow[hsel];
        bfrag.h[1] = brow[hsel + 2];

        // Unconditional speculative prefetch of this wave's B rows 4 tiles
        // ahead (TH=0 => speculative: OOB translation failures are silently
        // dropped per ISA, so no guard branch is needed in the hot loop).
        __builtin_prefetch(X2 + slice + (size_t)(k0 + 64 + lrow) * CC, 0, 1);

        v8f acc = {};
        v8f d = __builtin_amdgcn_wmma_f32_16x16x32_f16(
            /*neg_a=*/false, afrag.v, /*neg_b=*/false, bfrag.v,
            /*c_mod=*/(short)0, acc, /*reuse_a=*/false, /*reuse_b=*/false);

        // D layout: vgpr r -> M = r + 8*(lane>=16), N = lane%16
        const int col = k0 + lrow;
        const int rb  = hsel * 8;
#pragma unroll
        for (int r = 0; r < 8; ++r)
            logits[(rb + r) * NN + col] = d[r];
    }
    __syncthreads();

    // ---- softmax over k (1024) per row; 8 lanes per row, 4 rows per wave
    const int row = wv * 4 + (lane >> 3);   // 0..15
    const int seg = lane & 7;               // 0..7
    const float* lrowp = &logits[row * NN];

    float m = -INFINITY;
#pragma unroll 8
    for (int j = 0; j < NN / 8; ++j)
        m = fmaxf(m, lrowp[seg + 8 * j]);
    m = fmaxf(m, __shfl_xor(m, 1, 32));
    m = fmaxf(m, __shfl_xor(m, 2, 32));
    m = fmaxf(m, __shfl_xor(m, 4, 32));

    float sum = 0.0f;
#pragma unroll 8
    for (int j = 0; j < NN / 8; ++j)
        sum += __expf(lrowp[seg + 8 * j] - m);
    sum += __shfl_xor(sum, 1, 32);
    sum += __shfl_xor(sum, 2, 32);
    sum += __shfl_xor(sum, 4, 32);
    const float inv = 1.0f / sum;

    // adj[b, n0+row, k, s] : flat = ((b*N + n)*N + k)*S + s
    float* orow = out + (((size_t)b * NN + n0 + row) * NN) * SS + s;
#pragma unroll 4
    for (int j = 0; j < NN / 8; ++j) {
        const int k = seg + 8 * j;
        orow[(size_t)k * SS] = __expf(lrowp[k] - m) * inv;
    }
}

// ---------------------------------------------------------------------------
extern "C" void kernel_launch(void* const* d_in, const int* in_sizes, int n_in,
                              void* d_out, int out_size, void* d_ws, size_t ws_size,
                              hipStream_t stream)
{
    const float* x  = (const float*)d_in[0];
    const float* W1 = (const float*)d_in[1];
    const float* b1 = (const float*)d_in[2];
    const float* W2 = (const float*)d_in[3];
    const float* b2 = (const float*)d_in[4];
    float* out = (float*)d_out;

    const size_t elems = (size_t)BB * SS * NN * CC;   // 3,145,728 halves each
    _Float16* X1 = (_Float16*)d_ws;
    _Float16* X2 = X1 + elems;

    // Stage 1: channel projection + tanh  (B*N*S = 98304 positions)
    project_tanh_kernel<<<dim3((BB * NN * SS) / 256), 256, 0, stream>>>(
        x, W1, b1, W2, b2, X1, X2);

    // Stage 2: fused WMMA GEMM + softmax, one output pass
    gemm_softmax_kernel<<<dim3(NN / 16, SS, BB), 128, 0, stream>>>(X1, X2, out);
}
